// KnnUnpoolLayer_16484084483005
// MI455X (gfx1250) — matrix operand order repaired and checked by reading, not compiled, not run on hardware
//
#include <hip/hip_runtime.h>

#define D 128
#define CH 32            // database rows per LDS chunk (double-buffered)
#define PITCHH 264       // halves per LDS row: hi[0,128) lo[128,256) + 8 pad
#define TILE_HALVES (CH * PITCHH)      // 8448 halves = 16896 B per buffer
#define TILE_BYTES  (TILE_HALVES * 2)  // 16896

typedef _Float16 halfT;
typedef _Float16 v16h __attribute__((ext_vector_type(16)));
typedef float    v8f  __attribute__((ext_vector_type(8)));

// ---------------- kernel 1: copy x into out[0 : N*D) ----------------
__global__ __launch_bounds__(256)
void copy_x_kernel(const float4* __restrict__ src, float4* __restrict__ dst, int n4) {
    int i = blockIdx.x * 256 + threadIdx.x;
    if (i < n4) dst[i] = src[i];
}

// ---------------- kernel 2: xnorm[n] = ||x_n||^2 --------------------
__global__ __launch_bounds__(256)
void xnorm_kernel(const float* __restrict__ x, float* __restrict__ xnorm, int N) {
    int wave = (blockIdx.x * 256 + threadIdx.x) >> 5;   // one wave32 per row
    int lane = threadIdx.x & 31;
    if (wave >= N) return;
    const float4* row = (const float4*)(x + (size_t)wave * D);
    float4 v = row[lane];
    float s = v.x * v.x + v.y * v.y + v.z * v.z + v.w * v.w;
    #pragma unroll
    for (int off = 16; off > 0; off >>= 1) s += __shfl_xor(s, off, 32);
    if (lane == 0) xnorm[wave] = s;
}

// ----- kernel 2b: split x into packed f16 hi/lo planes in scratch ----
// row layout: 512B = [hi halves 0..127][lo halves 0..127]
__global__ __launch_bounds__(256)
void cvt_half_kernel(const float* __restrict__ x, halfT* __restrict__ xh, int total4) {
    int t = blockIdx.x * 256 + threadIdx.x;
    if (t >= total4) return;
    int row = t >> 5, c4 = t & 31;          // 32 float4 per row
    float4 v = *(const float4*)(x + (size_t)t * 4);
    halfT h0 = (halfT)v.x, h1 = (halfT)v.y, h2 = (halfT)v.z, h3 = (halfT)v.w;
    halfT l0 = (halfT)(v.x - (float)h0), l1 = (halfT)(v.y - (float)h1),
          l2 = (halfT)(v.z - (float)h2), l3 = (halfT)(v.w - (float)h3);
    union { halfT h[4]; uint2 u; } ph, pl;
    ph.h[0]=h0; ph.h[1]=h1; ph.h[2]=h2; ph.h[3]=h3;
    pl.h[0]=l0; pl.h[1]=l1; pl.h[2]=l2; pl.h[3]=l3;
    halfT* hd = xh + (size_t)row * 256 + c4 * 4;
    *(uint2*)hd         = ph.u;
    *(uint2*)(hd + 128) = pl.u;
}

// ------- kernel 3: fused split-f16 GEMM + top-3 + mean-gather -------
__global__ __launch_bounds__(256)
void knn_unpool_kernel(const float* __restrict__ x,
                       const int*   __restrict__ rand_inds,
                       const float* __restrict__ xnorm,
                       const halfT* __restrict__ xhalf,
                       float*       __restrict__ out,
                       int N) {
    // phase 1: two async-filled tile buffers, 2 x 16896 B = 8448 floats
    // phase 2: cand scores [8][16][16][3] (6144 f) + cand idx (6144 i)
    __shared__ __align__(16) float smem[12288];

    const int tid  = threadIdx.x;
    const int wave = tid >> 5;
    const int lane = tid & 31;
    const int lrow = lane & 15;
    const int lhi  = lane >> 4;
    const int m0   = (blockIdx.x * 8 + wave) * 16;   // query-tile base

    // ---- A fragments: 16x128 query tile split into hi/lo f16 planes ----
    const int qi = rand_inds[m0 + lrow];
    const float* arow = x + (size_t)qi * D;
    v16h aH[4], aL[4];
    #pragma unroll
    for (int s = 0; s < 4; ++s) {
        const float* p0 = arow + 32 * s + 8 * lhi;
        const float* p1 = p0 + 16;
        float4 u0 = *(const float4*)(p0);
        float4 u1 = *(const float4*)(p0 + 4);
        float4 u2 = *(const float4*)(p1);
        float4 u3 = *(const float4*)(p1 + 4);
        float vals[16] = {u0.x,u0.y,u0.z,u0.w, u1.x,u1.y,u1.z,u1.w,
                          u2.x,u2.y,u2.z,u2.w, u3.x,u3.y,u3.z,u3.w};
        #pragma unroll
        for (int h = 0; h < 16; ++h) {
            float v = vals[h];
            halfT hi = (halfT)v;
            halfT lo = (halfT)(v - (float)hi);
            aH[s][h] = hi;
            aL[s][h] = lo;
        }
    }

    // per-lane running top-3 (score,index) for rows r + 8*lhi
    float ts[8][3];
    int   ti[8][3];
    #pragma unroll
    for (int r = 0; r < 8; ++r) {
        ts[r][0] = ts[r][1] = ts[r][2] = 3.4e38f;
        ti[r][0] = ti[r][1] = ti[r][2] = 0;
    }

    // ---- async copy setup: 256 threads x 4 x b128 = 16 KB chunk ----
    // 8 threads per row; thread covers 64 contiguous bytes of one row
    const uint32_t voff0   = (uint32_t)tid * 64u;                       // src offset in chunk
    const uint32_t ldsrow  = (uint32_t)((tid >> 3) * (PITCHH * 2) + (tid & 7) * 64);
    const uint32_t lds0    = (uint32_t)(uintptr_t)(void*)smem;          // LDS byte offset

    const int nch = N / CH;

    // prologue: launch chunk 0 into buffer 0
    {
        uint64_t sb = (uint64_t)(uintptr_t)xhalf;
        #pragma unroll
        for (int j = 0; j < 4; ++j)
            asm volatile("global_load_async_to_lds_b128 %0, %1, %2"
                         :: "v"(lds0 + ldsrow + j * 16),
                            "v"(voff0 + j * 16),
                            "s"(sb) : "memory");
    }

    union V { uint4 q[2]; v16h v; };

    for (int c = 0; c < nch; ++c) {
        // my 4 async loads for chunk c have landed in LDS
        asm volatile("s_wait_asynccnt 0x0" ::: "memory");
        __syncthreads();   // everyone's chunk-c data landed; prev reads done

        // kick off chunk c+1 into the other buffer (overlaps compute below)
        if (c + 1 < nch) {
            uint64_t sb = (uint64_t)(uintptr_t)xhalf + (uint64_t)(c + 1) * (CH * 512);
            uint32_t dst = lds0 + (((c + 1) & 1) ? (uint32_t)TILE_BYTES : 0u) + ldsrow;
            #pragma unroll
            for (int j = 0; j < 4; ++j)
                asm volatile("global_load_async_to_lds_b128 %0, %1, %2"
                             :: "v"(dst + j * 16),
                                "v"(voff0 + j * 16),
                                "s"(sb) : "memory");
        }

        const halfT* tileh = (const halfT*)smem + (size_t)(c & 1) * TILE_HALVES;
        const int n0 = c * CH;

        #pragma unroll
        for (int sub = 0; sub < 2; ++sub) {
            const int colc = sub * 16 + lrow;            // this lane's column
            v8f acc = {};
            #pragma unroll
            for (int s = 0; s < 4; ++s) {
                // B fragments: 8 halves at K=32s+8*lhi, then 8 halves at K=32s+16+8*lhi
                const halfT* bp = &tileh[colc * PITCHH + 32 * s + 8 * lhi];
                V hb, lb;
                hb.q[0] = *(const uint4*)(bp);
                hb.q[1] = *(const uint4*)(bp + 16);
                lb.q[0] = *(const uint4*)(bp + 128);
                lb.q[1] = *(const uint4*)(bp + 144);
                acc = __builtin_amdgcn_wmma_f32_16x16x32_f16(
                        false, aL[s], false, hb.v, (short)0, acc, false, false);
                acc = __builtin_amdgcn_wmma_f32_16x16x32_f16(
                        false, aH[s], false, lb.v, (short)0, acc, false, false);
                acc = __builtin_amdgcn_wmma_f32_16x16x32_f16(
                        false, aH[s], false, hb.v, (short)0, acc, false, false);
            }
            const int   n  = n0 + colc;
            const float xn = xnorm[n];
            #pragma unroll
            for (int r = 0; r < 8; ++r) {
                float s = xn - 2.0f * acc[r];   // rank key (q-norm dropped)
                if (s < ts[r][2]) {             // strict < : earlier index wins ties
                    if (s < ts[r][1]) {
                        ts[r][2] = ts[r][1]; ti[r][2] = ti[r][1];
                        if (s < ts[r][0]) {
                            ts[r][1] = ts[r][0]; ti[r][1] = ti[r][0];
                            ts[r][0] = s;        ti[r][0] = n;
                        } else { ts[r][1] = s; ti[r][1] = n; }
                    } else { ts[r][2] = s; ti[r][2] = n; }
                }
            }
        }
    }

    // ---- merge 16 column-classes x top-3 per row via LDS ----
    __syncthreads();                       // done with tiles; reuse smem
    float* cs = smem;                      // [8][16][16][3] scores
    int*   ci = (int*)(smem + 6144);       // [8][16][16][3] indices
    #pragma unroll
    for (int r = 0; r < 8; ++r) {
        int row = r + 8 * lhi;
        int o = ((wave * 16 + row) * 16 + lrow) * 3;
        cs[o + 0] = ts[r][0]; cs[o + 1] = ts[r][1]; cs[o + 2] = ts[r][2];
        ci[o + 0] = ti[r][0]; ci[o + 1] = ti[r][1]; ci[o + 2] = ti[r][2];
    }
    __syncthreads();

    int f0 = 0, f1 = 0, f2 = 0;
    if (lane < 16) {                       // lane owns row `lane` of this tile
        float b0 = 3.4e38f, b1 = 3.4e38f, b2 = 3.4e38f;
        int   j0 = 0x7fffffff, j1 = 0x7fffffff, j2 = 0x7fffffff;
        int o = ((wave * 16 + lane) * 16) * 3;
        #pragma unroll 1
        for (int c = 0; c < 48; ++c) {
            float s = cs[o + c];
            int   j = ci[o + c];
            if ((s < b2) || (s == b2 && j < j2)) {
                if ((s < b1) || (s == b1 && j < j1)) {
                    b2 = b1; j2 = j1;
                    if ((s < b0) || (s == b0 && j < j0)) {
                        b1 = b0; j1 = j0; b0 = s; j0 = j;
                    } else { b1 = s; j1 = j; }
                } else { b2 = s; j2 = j; }
            }
        }
        f0 = j0; f1 = j1; f2 = j2;
    }
    __syncthreads();
    if (lane < 16) {
        ci[wave * 48 + lane * 3 + 0] = f0;
        ci[wave * 48 + lane * 3 + 1] = f1;
        ci[wave * 48 + lane * 3 + 2] = f2;
    }
    __syncthreads();

    // ---- gather 3 neighbor rows, mean, write out[N + m] ----
    const int c = lane * 4;
    const float third = 1.0f / 3.0f;
    #pragma unroll 1
    for (int r = 0; r < 16; ++r) {
        int i0 = ci[wave * 48 + r * 3 + 0];
        int i1 = ci[wave * 48 + r * 3 + 1];
        int i2 = ci[wave * 48 + r * 3 + 2];
        float4 a = *(const float4*)(x + (size_t)i0 * D + c);
        float4 b = *(const float4*)(x + (size_t)i1 * D + c);
        float4 g = *(const float4*)(x + (size_t)i2 * D + c);
        float4 o;
        o.x = (a.x + b.x + g.x) * third;
        o.y = (a.y + b.y + g.y) * third;
        o.z = (a.z + b.z + g.z) * third;
        o.w = (a.w + b.w + g.w) * third;
        *(float4*)(out + (size_t)(N + m0 + r) * D + c) = o;
    }
}

extern "C" void kernel_launch(void* const* d_in, const int* in_sizes, int n_in,
                              void* d_out, int out_size, void* d_ws, size_t ws_size,
                              hipStream_t stream) {
    const float* x         = (const float*)d_in[0];
    const int*   rand_inds = (const int*)d_in[1];
    float*       out       = (float*)d_out;
    float*       xnorm     = (float*)d_ws;                 // N floats
    halfT*       xhalf     = (halfT*)((float*)d_ws + in_sizes[0] / D);  // N*256 halves

    const int N = in_sizes[0] / D;                // 16384
    const int M = in_sizes[1];                    // 16384

    int n4 = N * (D / 4);
    copy_x_kernel<<<(n4 + 255) / 256, 256, 0, stream>>>((const float4*)x, (float4*)out, n4);
    xnorm_kernel<<<(N + 7) / 8, 256, 0, stream>>>(x, xnorm, N);
    cvt_half_kernel<<<(N * 32 + 255) / 256, 256, 0, stream>>>(x, xhalf, N * 32);
    knn_unpool_kernel<<<M / 128, 256, 0, stream>>>(x, rand_inds, xnorm, xhalf, out, N);
}
